// ORNN_59717225284167
// MI455X (gfx1250) — matrix operand order repaired and checked by reading, not compiled, not run on hardware
//
#include <hip/hip_runtime.h>
#include <cstdint>
#include <cstddef>

// ---------------------------------------------------------------------------
// Types for CDNA5 WMMA (wave32): v_wmma_f32_16x16x32_bf16
// ---------------------------------------------------------------------------
typedef __attribute__((ext_vector_type(16))) __bf16 v16bf;
typedef __attribute__((ext_vector_type(8)))  float  v8f;
typedef __attribute__((ext_vector_type(4)))  unsigned int v4u;

union Frag16 {
    v16bf v;
    v4u   q[2];
};

static __device__ __forceinline__ unsigned short f2bf(float f) {
    union { float f; unsigned u; } x; x.f = f;
    unsigned u = x.u;
    u += 0x7FFFu + ((u >> 16) & 1u);
    return (unsigned short)(u >> 16);
}

// A-fragment (16x32 bf16): lane<16 holds K {0..7,16..23}, lane>=16 {8..15,24..31}
static __device__ __forceinline__ void loadFragA(Frag16& f, const unsigned short* p) {
    f.q[0] = *(const v4u*)(p);
    f.q[1] = *(const v4u*)(p + 16);
}
// B-fragment (32x16 bf16): lane<16 holds K {0..7,16..23}, lane>=16 {8..15,24..31}
static __device__ __forceinline__ void loadFragB(Frag16& f, const unsigned short* p) {
    f.q[0] = *(const v4u*)(p);
    f.q[1] = *(const v4u*)(p + 16);
}
static __device__ __forceinline__ v8f wmma_bf16(const Frag16& a, const Frag16& b, v8f c) {
    return __builtin_amdgcn_wmma_f32_16x16x32_bf16(
        false, a.v, false, b.v, (short)0, c, false, false);
}

// ---------------------------------------------------------------------------
// fp32 -> bf16 conversion
// ---------------------------------------------------------------------------
__global__ __launch_bounds__(256) void k_f32_to_bf16(
    const float* __restrict__ in, unsigned short* __restrict__ out, int n) {
    int stride = gridDim.x * blockDim.x;
    for (int i = blockIdx.x * blockDim.x + threadIdx.x; i < n; i += stride)
        out[i] = f2bf(in[i]);
}

__global__ __launch_bounds__(256) void k_bias_sum(
    const float* __restrict__ a, const float* __restrict__ b,
    float* __restrict__ out, int n) {
    int i = blockIdx.x * blockDim.x + threadIdx.x;
    if (i < n) out[i] = a[i] + b[i];
}

// ---------------------------------------------------------------------------
// GEMM: out[m,n] = sum_k A[m,k]*W[n,k] + bias[n];  N = 512 fixed.
// 16K waves -> latency hidden by occupancy; fragments double-buffered.
// ---------------------------------------------------------------------------
template<int K>
__global__ __launch_bounds__(256) void k_gemm_bf16(
    const unsigned short* __restrict__ A,
    const unsigned short* __restrict__ W,
    const float* __restrict__ bias,
    float* __restrict__ out) {
    constexpr int N  = 512;
    constexpr int KB = K >> 5;
    const int lane   = threadIdx.x & 31;
    const int waveId = blockIdx.x * (blockDim.x >> 5) + (threadIdx.x >> 5);
    constexpr int nGroups = N >> 6;
    const int mTile = waveId / nGroups;
    const int nG    = waveId % nGroups;

    const int half = lane >> 4;
    const int l15  = lane & 15;

    const unsigned short* aRow  = A + (size_t)(mTile * 16 + l15) * K + half * 8;
    const unsigned short* wBase = W + (size_t)(nG * 64 + l15) * K + half * 8;
    const float*          bPtr  = bias + nG * 64 + l15;
    float*                oPtr  = out + (size_t)(mTile * 16 + half * 8) * N
                                      + nG * 64 + l15;

    v8f acc[4];
    #pragma unroll
    for (int j = 0; j < 4; ++j) {
        const float bn = bPtr[j * 16];
        #pragma unroll
        for (int r = 0; r < 8; ++r) acc[j][r] = bn;
    }

    Frag16 a0, b0[4];
    loadFragA(a0, aRow);
    #pragma unroll
    for (int j = 0; j < 4; ++j) loadFragB(b0[j], wBase + (size_t)j * 16 * K);

    #pragma unroll
    for (int kb = 0; kb < KB; ++kb) {
        Frag16 a1, b1[4];
        if (kb + 1 < KB) {
            loadFragA(a1, aRow + (kb + 1) * 32);
            #pragma unroll
            for (int j = 0; j < 4; ++j)
                loadFragB(b1[j], wBase + (size_t)j * 16 * K + (kb + 1) * 32);
        }
        #pragma unroll
        for (int j = 0; j < 4; ++j) acc[j] = wmma_bf16(a0, b0[j], acc[j]);
        a0 = a1;
        #pragma unroll
        for (int j = 0; j < 4; ++j) b0[j] = b1[j];
    }

    #pragma unroll
    for (int j = 0; j < 4; ++j)
        #pragma unroll
        for (int r = 0; r < 8; ++r)
            oPtr[(size_t)r * N + j * 16] = acc[j][r];
}

// ---------------------------------------------------------------------------
// Recurrent scan:  h = tanh(xw[:,t,:] + h @ Whh^T);  T = 512, H = 512.
// 4 WGs x 16 batch rows; 8 waves x 64 H-columns.
//  - h state in LDS (bf16, padded stride)
//  - 7/16 of Whh staged in LDS (fragment-swizzled, bank-conflict-free,
//    286.7KB) -> 7 of 16 B-fragment K-blocks come from DS, 9 from L2.  Total LDS 303.4KB.
//  - t-loop unrolled x2 with ping-pong accumulator banks (xw(t+1) loads land
//    directly in the spare bank under step t's WMMAs).
//  - layer-0 output emitted as a coalesced hbuf->global b128 copy.
// ---------------------------------------------------------------------------
#define HB_LD 520   // dword stride 260, gcd(260,64)=4 -> conflict-free A reads

template<bool LAST>
__global__ __launch_bounds__(256) void k_rnn_scan(
    const float* __restrict__ xw,            // [(b*T + t)*H + n] fp32
    const unsigned short* __restrict__ Whh,  // [H,H] bf16 row-major
    const float* __restrict__ h0,            // [B,H] fp32 (this layer)
    unsigned short* __restrict__ outBf,      // bf16 (LAST=false)
    float* __restrict__ outF,                // fp32 (LAST=true)
    float* __restrict__ hn) {                // [B,H] fp32 last hidden
    constexpr int T = 512, H = 512, KB = 16;
    constexpr int WKB  = 7;    // K-blocks of Whh staged in LDS
    constexpr int WCOL = 40;   // padded shorts per (kb,column): 80B, 16B-aligned
    constexpr size_t TH = (size_t)T * H;
    extern __shared__ unsigned short smem[];
    unsigned short* hbuf = smem;                    // 16*HB_LD shorts
    unsigned short* wlds = smem + 16 * HB_LD;       // WKB*512*WCOL shorts

    const int tid   = threadIdx.x;
    const int lane  = tid & 31;
    const int wave  = tid >> 5;
    const int half  = lane >> 4;
    const int l15   = lane & 15;
    const int btile = blockIdx.x;
    const int ml0   = half * 8;

    // stage Whh K-blocks 0..WKB-1 into LDS (fragment-swizzled)
    for (int i = tid; i < WKB * 512 * 4; i += 256) {
        const int p  = i & 3;            // 16B chunk within 32 shorts
        const int n  = (i >> 2) & 511;
        const int kb = i >> 11;
        v4u d = *(const v4u*)&Whh[(size_t)n * H + kb * 32 + p * 8];
        *(v4u*)&wlds[(size_t)(kb * 512 + n) * WCOL + p * 8] = d;
    }
    // initial hidden state
    for (int i = tid; i < 16 * H; i += blockDim.x) {
        const int mm = i >> 9;
        const int nn = i & 511;
        hbuf[mm * HB_LD + nn] = f2bf(h0[(size_t)(btile * 16 + mm) * H + nn]);
    }
    __syncthreads();

    const int n0 = wave * 64 + l15;
    const unsigned short* aBase    = &hbuf[l15 * HB_LD + half * 8];
    const unsigned short* wBase    = Whh  + (size_t)n0 * H + half * 8;
    const unsigned short* wldsBase = wlds + (size_t)n0 * WCOL + half * 8;

    const size_t rowBase = (size_t)(btile * 16 + ml0) * T * H + n0;
    const float*    xwPtr = xw + rowBase;            // advance by H per step
    float*          ofPtr = LAST ? outF + rowBase : nullptr;
    float*          hnPtr = hn + (size_t)(btile * 16 + ml0) * H + n0;
    unsigned short* hRow  = &hbuf[ml0 * HB_LD + n0];

    // coalesced layer-0 output copy: thread -> (row0 + 4c, col0..col0+7)
    const int cRow = tid >> 6;            // 0..3
    const int cCol = (tid & 63) * 8;      // 0..504
    unsigned short* cpDst = LAST ? nullptr
        : outBf + ((size_t)(btile * 16 + cRow) * T) * H + cCol;
    const unsigned short* cpSrc = &hbuf[cRow * HB_LD + cCol];

    auto loadXW = [&](v8f (&acc)[4], const float* p) {
        #pragma unroll
        for (int j = 0; j < 4; ++j)
            #pragma unroll
            for (int r = 0; r < 8; ++r)
                acc[j][r] = p[(size_t)r * TH + j * 16];
    };

    auto loadB = [&](int kb, Frag16 (&b)[4]) {
        if (kb < WKB) {
            #pragma unroll
            for (int j = 0; j < 4; ++j)
                loadFragB(b[j], wldsBase + (size_t)(kb * 512 + j * 16) * WCOL);
        } else {
            #pragma unroll
            for (int j = 0; j < 4; ++j)
                loadFragB(b[j], wBase + (size_t)j * 16 * H + kb * 32);
        }
    };

    auto gemmStep = [&](v8f (&acc)[4]) {
        Frag16 a0, b0[4];
        loadFragA(a0, aBase);
        loadB(0, b0);
        #pragma unroll
        for (int kb = 0; kb < KB; ++kb) {
            Frag16 a1, b1[4];
            if (kb + 1 < KB) {
                loadFragA(a1, aBase + (kb + 1) * 32);
                loadB(kb + 1, b1);
            }
            #pragma unroll
            for (int j = 0; j < 4; ++j) acc[j] = wmma_bf16(a0, b0[j], acc[j]);
            a0 = a1;
            #pragma unroll
            for (int j = 0; j < 4; ++j) b0[j] = b1[j];
        }
    };

    auto writeback = [&](v8f (&acc)[4], int t) {
        __syncthreads();   // all hbuf reads for this step complete
        #pragma unroll
        for (int j = 0; j < 4; ++j) {
            #pragma unroll
            for (int r = 0; r < 8; ++r) {
                const float v = tanhf(acc[j][r]);
                hRow[r * HB_LD + j * 16] = f2bf(v);
                if (LAST)
                    __builtin_nontemporal_store(v, ofPtr + (size_t)r * TH + j * 16);
                if (t == T - 1) hnPtr[(size_t)r * H + j * 16] = v;
            }
        }
        __syncthreads();   // new state visible to all
        if (LAST) {
            ofPtr += H;
        } else {
            // coalesced bf16 output: values identical to hbuf contents
            #pragma unroll
            for (int c = 0; c < 4; ++c) {
                v4u d = *(const v4u*)(cpSrc + (size_t)c * 4 * HB_LD);
                *(v4u*)(cpDst + (size_t)c * 4 * TH) = d;
            }
            cpDst += H;
        }
    };

    v8f accA[4], accB[4];
    loadXW(accA, xwPtr);
    for (int t = 0; t < T; t += 2) {
        loadXW(accB, xwPtr + H);               // step t+1 (T even)
        gemmStep(accA);
        writeback(accA, t);
        if (t + 2 < T) loadXW(accA, xwPtr + 2 * H);
        gemmStep(accB);
        writeback(accB, t + 1);
        xwPtr += 2 * H;
    }
}

// ---------------------------------------------------------------------------
// Host launcher
// ---------------------------------------------------------------------------
extern "C" void kernel_launch(void* const* d_in, const int* in_sizes, int n_in,
                              void* d_out, int out_size, void* d_ws, size_t ws_size,
                              hipStream_t stream) {
    (void)in_sizes; (void)n_in; (void)out_size; (void)ws_size;
    const int B = 64, T = 512, I = 256, H = 512;
    const size_t M = (size_t)B * T;
    constexpr size_t SCAN_LDS = (16 * 520 + 7 * 512 * 40) * sizeof(unsigned short);

    const float* x    = (const float*)d_in[0];
    const float* h0   = (const float*)d_in[1];
    const float* Wih0 = (const float*)d_in[2];
    const float* Whh0 = (const float*)d_in[3];
    const float* bih0 = (const float*)d_in[4];
    const float* bhh0 = (const float*)d_in[5];
    const float* Wih1 = (const float*)d_in[6];
    const float* Whh1 = (const float*)d_in[7];
    const float* bih1 = (const float*)d_in[8];
    const float* bhh1 = (const float*)d_in[9];

    float* out1 = (float*)d_out;
    float* hn   = out1 + (size_t)B * T * H;

    uint8_t* ws = (uint8_t*)d_ws;
    float* xW = (float*)ws;                                   // M*H fp32 (reused)
    uint8_t* regA = ws + M * H * sizeof(float);
    unsigned short* xbf    = (unsigned short*)regA;           // M*I bf16
    unsigned short* out0bf = (unsigned short*)regA;           // M*H bf16 (overlap)
    unsigned short* wih0b = (unsigned short*)(regA + M * H * sizeof(unsigned short));
    unsigned short* whh0b = wih0b + (size_t)H * I;
    unsigned short* wih1b = whh0b + (size_t)H * H;
    unsigned short* whh1b = wih1b + (size_t)H * H;
    float* b0 = (float*)(whh1b + (size_t)H * H);
    float* b1 = b0 + H;

    // allow >64KB dynamic LDS for the scan kernels (320KB/WGP on CDNA5)
    (void)hipFuncSetAttribute((const void*)&k_rnn_scan<false>,
                              hipFuncAttributeMaxDynamicSharedMemorySize,
                              (int)SCAN_LDS);
    (void)hipFuncSetAttribute((const void*)&k_rnn_scan<true>,
                              hipFuncAttributeMaxDynamicSharedMemorySize,
                              (int)SCAN_LDS);

    k_f32_to_bf16<<<2048, 256, 0, stream>>>(x,    xbf,   B * T * I);
    k_f32_to_bf16<<<256,  256, 0, stream>>>(Wih0, wih0b, H * I);
    k_f32_to_bf16<<<512,  256, 0, stream>>>(Whh0, whh0b, H * H);
    k_f32_to_bf16<<<512,  256, 0, stream>>>(Wih1, wih1b, H * H);
    k_f32_to_bf16<<<512,  256, 0, stream>>>(Whh1, whh1b, H * H);
    k_bias_sum<<<2, 256, 0, stream>>>(bih0, bhh0, b0, H);
    k_bias_sum<<<2, 256, 0, stream>>>(bih1, bhh1, b1, H);

    const int gemmBlocks = (int)((M / 16) * (H / 64) / 8);

    // layer 0
    k_gemm_bf16<256><<<gemmBlocks, 256, 0, stream>>>(xbf, wih0b, b0, xW);
    k_rnn_scan<false><<<4, 256, SCAN_LDS, stream>>>(xW, whh0b, h0, out0bf,
                                                    (float*)nullptr, hn);
    // layer 1
    k_gemm_bf16<512><<<gemmBlocks, 256, 0, stream>>>(out0bf, wih1b, b1, xW);
    k_rnn_scan<true><<<4, 256, SCAN_LDS, stream>>>(xW, whh1b, h0 + (size_t)B * H,
                                                   (unsigned short*)nullptr, out1,
                                                   hn + (size_t)B * H);
}
